// DynamicReductionNetworkObject_4535485464634
// MI455X (gfx1250) — matrix operand order repaired and checked by reading, not compiled, not run on hardware
//
#include <hip/hip_runtime.h>
#include <hip/hip_bf16.h>

// ---------------------------------------------------------------------------
// DynamicReductionNetwork (DGCNN-style) for MI455X / gfx1250 (wave32, WMMA).
// B=8, N=3072, IN_DIM=3, HD=64, K=16, L=2.
// - fused WMMA distance + top-K (never materialize the 302MB d2 matrix)
// - f16 WMMA w/ f32 accumulate for distance GEMM and edge MLPs
// - async global->LDS (ASYNCcnt) weight staging in the edge MLP
// - everything L2 resident (h = 6MB << 192MB L2)
// ---------------------------------------------------------------------------

#define BATCH 8
#define NPTS  3072
#define HD    64
#define KNN   16
#define NLAYER 2

typedef __attribute__((ext_vector_type(16))) _Float16 v16h;
typedef __attribute__((ext_vector_type(8)))  float    v8f;

__device__ __forceinline__ float elu_f(float x) {
    return x > 0.0f ? x : (__expf(x) - 1.0f);
}

// gfx1250 async copy: global -> LDS, 16B per lane, tracked by ASYNCcnt.
__device__ __forceinline__ void async_cp_b128(unsigned int lds_addr,
                                              unsigned long long gaddr) {
    asm volatile("global_load_async_to_lds_b128 %0, %1, off"
                 :: "v"(lds_addr), "v"(gaddr) : "memory");
}
__device__ __forceinline__ void wait_async0() {
    asm volatile("s_wait_asynccnt 0" ::: "memory");
}

// ---------------------------------------------------------------------------
// Convert W1/W2 (f32, [l][k][n]) to f16 transposed [l][n][k] so that WMMA
// B-fragments are contiguous 16-half (32B) loads.
// ---------------------------------------------------------------------------
__global__ void convert_weights_kernel(const float* __restrict__ W1,
                                       const float* __restrict__ W2,
                                       _Float16* __restrict__ W1t,
                                       _Float16* __restrict__ W2t) {
    int e = blockIdx.x * blockDim.x + threadIdx.x;
    if (e < NLAYER * 128 * 128) {
        int l = e / (128 * 128), r = e % (128 * 128);
        int n = r / 128, k = r % 128;
        W1t[e] = (_Float16)W1[l * 128 * 128 + k * 128 + n];
    }
    int e2 = e - NLAYER * 128 * 128;
    if (e2 >= 0 && e2 < NLAYER * 64 * 128) {
        int l = e2 / (64 * 128), r = e2 % (64 * 128);
        int n = r / 128, k = r % 128;
        W2t[e2] = (_Float16)W2[l * 128 * 64 + k * 64 + n];
    }
}

// ---------------------------------------------------------------------------
// h = elu(x * datanorm @ W_in + b_in); also emit f16 copy and row norms.
// ---------------------------------------------------------------------------
__global__ void embed_kernel(const float* __restrict__ x,
                             const float* __restrict__ dn,
                             const float* __restrict__ Win,
                             const float* __restrict__ bin,
                             float* __restrict__ h,
                             _Float16* __restrict__ hH,
                             float* __restrict__ xx) {
    int p = blockIdx.x * blockDim.x + threadIdx.x;
    if (p >= BATCH * NPTS) return;
    float x0 = x[p * 3 + 0] * dn[0];
    float x1 = x[p * 3 + 1] * dn[1];
    float x2 = x[p * 3 + 2] * dn[2];
    float s = 0.0f;
    for (int d = 0; d < HD; ++d) {
        float v = x0 * Win[d] + x1 * Win[64 + d] + x2 * Win[128 + d] + bin[d];
        v = elu_f(v);
        h[(size_t)p * HD + d]  = v;
        hH[(size_t)p * HD + d] = (_Float16)v;
        s += v * v;
    }
    xx[p] = s;
}

// Re-derive the f16 copy + norms from an updated h buffer.
__global__ void refresh_kernel(const float* __restrict__ h,
                               _Float16* __restrict__ hH,
                               float* __restrict__ xx) {
    int p = blockIdx.x * blockDim.x + threadIdx.x;
    if (p >= BATCH * NPTS) return;
    const float* r = h + (size_t)p * HD;
    float s = 0.0f;
    for (int d = 0; d < HD; ++d) {
        float v = r[d];
        s += v * v;
        hH[(size_t)p * HD + d] = (_Float16)v;
    }
    xx[p] = s;
}

// ---------------------------------------------------------------------------
// Fused kNN: block handles one batch b and a 16-row tile of query points.
// 8 waves split the 3072 candidate columns (24 tiles of 16 each).
// Each 16x16 dot-product tile = 2 chained v_wmma_f32_16x16x32_f16 (K=64).
// The d2 tile is produced and consumed by the same wave (DS ops of one wave
// are in-order, ISA 7.3), so no block barrier is needed inside the tile loop.
// Per-lane register top-16 with a cached worst-entry threshold: a rejected
// candidate costs one compare; the 16-wide rescan runs only on inserts.
// ---------------------------------------------------------------------------
__global__ void knn_kernel(const _Float16* __restrict__ hH,
                           const float* __restrict__ xx,
                           int* __restrict__ idxout) {
    __shared__ _Float16 sA[16 * HD];        // query tile, f16
    __shared__ float    sXXi[16];
    __shared__ float    sD[8][16][17];      // per-wave d2 tiles (padded rows)
    __shared__ float    sMd[16][128];       // merge buffer (8 waves x 16)
    __shared__ int      sMi[16][128];

    const int bN   = NPTS / 16;             // 192 tiles per batch
    const int b    = blockIdx.x / bN;
    const int i0   = (blockIdx.x % bN) * 16;
    const int tid  = threadIdx.x;
    const int w    = tid >> 5;
    const int lane = tid & 31;

    const _Float16* hb = hH + (size_t)b * NPTS * HD;

    // stage the 16x64 query tile in LDS
    for (int e = tid; e < 16 * HD; e += 256) {
        int r = e >> 6, k = e & 63;
        sA[e] = hb[(size_t)(i0 + r) * HD + k];
    }
    if (tid < 16) sXXi[tid] = xx[b * NPTS + i0 + tid];
    __syncthreads();

    const int nlane = lane & 15;            // column / row-in-tile owned
    const int koff  = (lane >> 4) ? 16 : 0; // A-fragment K-offset
    const int khalf = (lane >> 4) << 4;     // B-fragment K-half

    // Build the two A fragments (K = 0..31, 32..63) once.
    v16h afrag[2];
#pragma unroll
    for (int c = 0; c < 2; ++c) {
        int kbase = 32 * c;
#pragma unroll
        for (int e = 0; e < 16; ++e) {
            int k = kbase + koff + e;
            afrag[c][e] = sA[nlane * HD + k];
        }
    }

    float bestd[KNN];
    int   besti[KNN];
#pragma unroll
    for (int q = 0; q < KNN; ++q) { bestd[q] = 3.0e38f; besti[q] = 0; }
    float mx = 3.0e38f;                     // cached worst entry + its slot
    int   mq = 0;

    const int TILES = NPTS / (8 * 16);      // 24 per wave
#pragma unroll 1
    for (int t = 0; t < TILES; ++t) {
        int j0 = (t * 8 + w) * 16;

        // prefetch next candidate tile into cache (global_prefetch_b8)
        if (t + 1 < TILES) {
            __builtin_prefetch(hb + (size_t)(j0 + 128 + nlane) * HD, 0, 1);
        }

        // dot(i, j) tile via two chained WMMAs
        v8f acc = {};
#pragma unroll
        for (int c = 0; c < 2; ++c) {
            int kbase = 32 * c;
            v16h bfrag = *(const v16h*)(hb + (size_t)(j0 + nlane) * HD + kbase + khalf);
            acc = __builtin_amdgcn_wmma_f32_16x16x32_f16(
                false, afrag[c], false, bfrag, (short)0, acc, false, false);
        }

        // d2 tile -> this wave's LDS region
        float xxj   = xx[b * NPTS + j0 + nlane];
        int   mbase = (lane >> 4) * 8;
#pragma unroll
        for (int r = 0; r < 8; ++r) {
            int m = mbase + r;
            sD[w][m][nlane] = sXXi[m] + xxj - 2.0f * acc[r];
        }

        // lanes 0..15 own one query row each: pull the whole row into
        // registers behind a single DS wait, then scan fully unrolled.
        if (lane < 16) {
            const float* rowp = &sD[w][lane][0];
            float rv[16];
#pragma unroll
            for (int n = 0; n < 16; ++n) rv[n] = rowp[n];
#pragma unroll
            for (int n = 0; n < 16; ++n) {
                float d = rv[n];
                if (d < mx) {               // fast reject: 1 compare
                    int jj = j0 + n;
#pragma unroll
                    for (int q = 0; q < KNN; ++q)
                        if (q == mq) { bestd[q] = d; besti[q] = jj; }
                    // rescan for the new worst entry (insert path only)
                    mx = bestd[0]; mq = 0;
#pragma unroll
                    for (int q = 1; q < KNN; ++q)
                        if (bestd[q] > mx) { mx = bestd[q]; mq = q; }
                }
            }
        }
    }

    // cross-wave merge: 8 partial top-16 lists per row -> final top-16
    if (lane < 16) {
#pragma unroll
        for (int q = 0; q < KNN; ++q) {
            sMd[lane][w * 16 + q] = bestd[q];
            sMi[lane][w * 16 + q] = besti[q];
        }
    }
    __syncthreads();
    if (w == 0 && lane < 16) {
        float fd[KNN]; int fi[KNN];
#pragma unroll
        for (int q = 0; q < KNN; ++q) { fd[q] = 3.0e38f; fi[q] = 0; }
        float fmx = 3.0e38f;
        int   fmq = 0;
#pragma unroll 1
        for (int g = 0; g < 32; ++g) {      // 128 candidates, 4 per LDS wait
            float4 dv = *(const float4*)&sMd[lane][g * 4];
            int4   iv = *(const int4*)&sMi[lane][g * 4];
            float dd[4] = { dv.x, dv.y, dv.z, dv.w };
            int   ii[4] = { iv.x, iv.y, iv.z, iv.w };
#pragma unroll
            for (int u = 0; u < 4; ++u) {
                float d = dd[u];
                if (d < fmx) {
                    int jj = ii[u];
#pragma unroll
                    for (int q = 0; q < KNN; ++q)
                        if (q == fmq) { fd[q] = d; fi[q] = jj; }
                    fmx = fd[0]; fmq = 0;
#pragma unroll
                    for (int q = 1; q < KNN; ++q)
                        if (fd[q] > fmx) { fmx = fd[q]; fmq = q; }
                }
            }
        }
#pragma unroll
        for (int q = 0; q < KNN; ++q)
            idxout[((size_t)b * NPTS + i0 + lane) * KNN + q] = fi[q];
    }
}

// ---------------------------------------------------------------------------
// Edge MLP: one wave per point. Its 16 neighbors are the 16 M-rows of the
// WMMA tiles. msg(128) @ W1(128x128) -> elu -> @ W2(128x64) -> elu -> sum_k.
// W1t/W2t staged in LDS via GLOBAL_LOAD_ASYNC_TO_LDS_B128 (overlapped with
// the per-wave neighbor gather / A-fragment build). ~81KB dynamic LDS.
// ---------------------------------------------------------------------------
__global__ void edge_mlp_kernel(const float* __restrict__ h,
                                const int* __restrict__ idx,
                                const _Float16* __restrict__ W1t,
                                const float* __restrict__ b1,
                                const _Float16* __restrict__ W2t,
                                const float* __restrict__ b2,
                                float* __restrict__ hout) {
    extern __shared__ __align__(32) char smem[];
    _Float16* sW1 = (_Float16*)smem;                    // 128*128 f16 (32KB)
    _Float16* sW2 = sW1 + 128 * 128;                    // 64*128  f16 (16KB)
    float*    sB1 = (float*)(sW2 + 64 * 128);           // 128 f32
    float*    sB2 = sB1 + 128;                          // 64 f32
    _Float16* sM1 = (_Float16*)(sB2 + 64);              // 8 waves * 16*128 f16

    const int tid  = threadIdx.x;
    const int w    = tid >> 5;
    const int lane = tid & 31;

    // --- async stage W1t (2048 x 16B) + W2t (1024 x 16B) into LDS ----------
    {
        unsigned int       l1 = (unsigned int)(uintptr_t)sW1;
        unsigned long long g1 = (unsigned long long)(uintptr_t)W1t;
#pragma unroll
        for (int e = 0; e < 8; ++e) {       // 2048 chunks / 256 threads
            int c = tid + e * 256;
            async_cp_b128(l1 + c * 16, g1 + (unsigned long long)c * 16);
        }
        unsigned int       l2 = (unsigned int)(uintptr_t)sW2;
        unsigned long long g2 = (unsigned long long)(uintptr_t)W2t;
#pragma unroll
        for (int e = 0; e < 4; ++e) {       // 1024 chunks / 256 threads
            int c = tid + e * 256;
            async_cp_b128(l2 + c * 16, g2 + (unsigned long long)c * 16);
        }
    }
    if (tid < 128) sB1[tid] = b1[tid];
    if (tid < 64)  sB2[tid] = b2[tid];

    const int p = blockIdx.x * 8 + w;       // global point id
    const int b = p / NPTS;
    const int i = p % NPTS;
    const float* hb = h + (size_t)b * NPTS * HD;

    const int nrow  = lane & 15;            // neighbor row / output column
    const int koff  = (lane >> 4) ? 16 : 0;
    const int khalf = (lane >> 4) << 4;
    const int mbase = (lane >> 4) * 8;

    const int jn = idx[(size_t)p * KNN + nrow];
    const float* xi = hb + (size_t)i * HD;
    const float* xj = hb + (size_t)jn * HD;

    // A fragments for GEMM1: msg = [xi (f<64), xj-xi (f>=64)], built in regs
    // while the async weight staging is in flight.
    v16h a1[4];
#pragma unroll
    for (int c = 0; c < 4; ++c) {
        int kbase = 32 * c;
#pragma unroll
        for (int e = 0; e < 16; ++e) {
            int f = kbase + koff + e;
            float v = (f < 64) ? xi[f] : (xj[f - 64] - xi[f - 64]);
            a1[c][e] = (_Float16)v;
        }
    }

    wait_async0();                          // s_wait_asynccnt 0
    __syncthreads();

    _Float16* myM1 = sM1 + (size_t)w * 16 * 128;

    // GEMM1: 16x128 @ 128x128, elu, stage f16
#pragma unroll
    for (int t = 0; t < 8; ++t) {
        v8f acc = {};
#pragma unroll
        for (int c = 0; c < 4; ++c) {
            const _Float16* bp = sW1 + (size_t)(16 * t + nrow) * 128 + 32 * c + khalf;
            v16h bfrag = *(const v16h*)bp;
            acc = __builtin_amdgcn_wmma_f32_16x16x32_f16(
                false, a1[c], false, bfrag, (short)0, acc, false, false);
        }
        int   ncol = 16 * t + nrow;
        float bias = sB1[ncol];
#pragma unroll
        for (int r = 0; r < 8; ++r) {
            float v = elu_f(acc[r] + bias);
            myM1[(size_t)(mbase + r) * 128 + ncol] = (_Float16)v;
        }
    }
    __syncthreads();

    // A fragments for GEMM2 from staged intermediate
    v16h a2[4];
#pragma unroll
    for (int c = 0; c < 4; ++c) {
        int kbase = 32 * c;
#pragma unroll
        for (int e = 0; e < 16; ++e) {
            int f = kbase + koff + e;
            a2[c][e] = myM1[(size_t)nrow * 128 + f];
        }
    }

    // GEMM2: 16x128 @ 128x64, elu, sum over the 16 neighbor rows
#pragma unroll
    for (int t = 0; t < 4; ++t) {
        v8f acc = {};
#pragma unroll
        for (int c = 0; c < 4; ++c) {
            const _Float16* bp = sW2 + (size_t)(16 * t + nrow) * 128 + 32 * c + khalf;
            v16h bfrag = *(const v16h*)bp;
            acc = __builtin_amdgcn_wmma_f32_16x16x32_f16(
                false, a2[c], false, bfrag, (short)0, acc, false, false);
        }
        int   ncol = 16 * t + nrow;
        float bias = sB2[ncol];
        float s = 0.0f;
#pragma unroll
        for (int r = 0; r < 8; ++r) s += elu_f(acc[r] + bias);
        s += __shfl_xor(s, 16, 32);         // combine rows 0..7 with 8..15
        if (lane < 16)
            hout[((size_t)b * NPTS + i) * HD + ncol] = s;
    }
}

// ---------------------------------------------------------------------------
// Max-pool over N, then the tiny 64->64->64->1 head. Grid = B, block = 64.
// ---------------------------------------------------------------------------
__global__ void pool_head_kernel(const float* __restrict__ h,
                                 const float* __restrict__ Wo1, const float* __restrict__ bo1,
                                 const float* __restrict__ Wo2, const float* __restrict__ bo2,
                                 const float* __restrict__ Wo3, const float* __restrict__ bo3,
                                 float* __restrict__ out) {
    __shared__ float sP[64], s1[64], s2[64];
    int b = blockIdx.x, d = threadIdx.x;
    float m = -3.0e38f;
    for (int n = 0; n < NPTS; ++n)
        m = fmaxf(m, h[((size_t)b * NPTS + n) * HD + d]);
    sP[d] = m;
    __syncthreads();
    float a = bo1[d];
    for (int k = 0; k < 64; ++k) a += sP[k] * Wo1[k * 64 + d];
    s1[d] = elu_f(a);
    __syncthreads();
    float c = bo2[d];
    for (int k = 0; k < 64; ++k) c += s1[k] * Wo2[k * 64 + d];
    s2[d] = elu_f(c);
    __syncthreads();
    if (d == 0) {
        float o = bo3[0];
        for (int k = 0; k < 64; ++k) o += s2[k] * Wo3[k];
        out[b] = o;
    }
}

// ---------------------------------------------------------------------------
extern "C" void kernel_launch(void* const* d_in, const int* in_sizes, int n_in,
                              void* d_out, int out_size, void* d_ws, size_t ws_size,
                              hipStream_t stream) {
    (void)in_sizes; (void)n_in; (void)out_size; (void)ws_size;

    const float* x   = (const float*)d_in[0];
    const float* dn  = (const float*)d_in[1];
    const float* Win = (const float*)d_in[2];
    const float* bin = (const float*)d_in[3];
    const float* W1  = (const float*)d_in[4];
    const float* b1  = (const float*)d_in[5];
    const float* W2  = (const float*)d_in[6];
    const float* b2  = (const float*)d_in[7];
    const float* Wo1 = (const float*)d_in[8];
    const float* bo1 = (const float*)d_in[9];
    const float* Wo2 = (const float*)d_in[10];
    const float* bo2 = (const float*)d_in[11];
    const float* Wo3 = (const float*)d_in[12];
    const float* bo3 = (const float*)d_in[13];

    char* ws = (char*)d_ws;
    size_t off = 0;
    float* hA = (float*)(ws + off);      off += (size_t)BATCH * NPTS * HD * 4;
    float* hB = (float*)(ws + off);      off += (size_t)BATCH * NPTS * HD * 4;
    _Float16* hH = (_Float16*)(ws + off); off += (size_t)BATCH * NPTS * HD * 2;
    float* xx = (float*)(ws + off);      off += (size_t)BATCH * NPTS * 4;
    int* nidx = (int*)(ws + off);        off += (size_t)BATCH * NPTS * KNN * 4;
    _Float16* W1t = (_Float16*)(ws + off); off += (size_t)NLAYER * 128 * 128 * 2;
    _Float16* W2t = (_Float16*)(ws + off); off += (size_t)NLAYER * 64 * 128 * 2;

    const int npoints = BATCH * NPTS;
    const size_t smemEdge = (size_t)128 * 128 * 2 + 64 * 128 * 2 + 128 * 4 + 64 * 4
                          + (size_t)8 * 16 * 128 * 2;   // ~81.7 KB (< 320KB WGP LDS)

    convert_weights_kernel<<<192, 256, 0, stream>>>(W1, W2, W1t, W2t);
    embed_kernel<<<(npoints + 255) / 256, 256, 0, stream>>>(x, dn, Win, bin, hA, hH, xx);

    // layer 0
    knn_kernel<<<BATCH * (NPTS / 16), 256, 0, stream>>>(hH, xx, nidx);
    edge_mlp_kernel<<<npoints / 8, 256, smemEdge, stream>>>(
        hA, nidx, W1t, b1, W2t, b2, hB);
    refresh_kernel<<<(npoints + 255) / 256, 256, 0, stream>>>(hB, hH, xx);

    // layer 1
    knn_kernel<<<BATCH * (NPTS / 16), 256, 0, stream>>>(hH, xx, nidx);
    edge_mlp_kernel<<<npoints / 8, 256, smemEdge, stream>>>(
        hB, nidx, W1t + 128 * 128, b1 + 128, W2t + 64 * 128, b2 + 64, hA);

    pool_head_kernel<<<BATCH, 64, 0, stream>>>(hA, Wo1, bo1, Wo2, bo2, Wo3, bo3,
                                               (float*)d_out);
}
